// MolmoAttention_31842887532708
// MI455X (gfx1250) — compile-verified, matrix operand order checked by reading, not run on hardware
//
#include <hip/hip_runtime.h>
#include <hip/hip_bf16.h>

typedef __attribute__((ext_vector_type(16))) __bf16 v16bf;
typedef __attribute__((ext_vector_type(8)))  float  v8f;

#define T_SEQ 2048
#define HID   4096
#define NH    32
#define HD    128
#define QKV_N 12288   // HID + 2*KV_SIZE

__device__ __forceinline__ unsigned short f32_to_bf16(float f) {
  unsigned int u = __float_as_uint(f);
  u += 0x7FFFu + ((u >> 16) & 1u);           // round-to-nearest-even
  return (unsigned short)(u >> 16);
}

union FragBF {
  v16bf v;
  unsigned short u[16];
  unsigned int d[8];
};

// K-dim index held by element i of a lane's 16-bit A/B fragment (ISA 7.12.2)
__device__ __forceinline__ int kmap(int lane, int i) {
  return ((i & 8) << 1) | ((lane & 16) >> 1) | (i & 7);
}
__device__ __forceinline__ int kdw(int lane, int p) {
  return kmap(lane, 2 * p) >> 1;
}

__device__ __forceinline__ v8f wmma_bf16(v16bf a, v16bf b, v8f c) {
  return __builtin_amdgcn_wmma_f32_16x16x32_bf16(false, a, false, b,
                                                 (short)0, c, false, false);
}

// ---- CDNA5 async DMA: global memory -> LDS, tracked by ASYNCcnt -----------
__device__ __forceinline__ void async_load_b128(const void* gaddr, void* lds) {
  unsigned l = (unsigned)(uintptr_t)lds;   // LDS aperture: addr[31:0] is LDS offset
  asm volatile("global_load_async_to_lds_b128 %0, %1, off"
               :: "v"(l), "v"(gaddr) : "memory");
}
__device__ __forceinline__ void wait_async0() {
  asm volatile("s_wait_asynccnt 0x0" ::: "memory");
}

// ---------------------------------------------------------------- f32 -> bf16
__global__ __launch_bounds__(256) void cvt_bf16(const float* __restrict__ src,
                                                unsigned short* __restrict__ dst,
                                                int n) {
  int stride = gridDim.x * blockDim.x;
  for (int i = blockIdx.x * blockDim.x + threadIdx.x; i < n; i += stride)
    dst[i] = f32_to_bf16(src[i]);
}

// ------------------------------- f32 [R][C] -> bf16 transposed [C][R] (tiled)
__global__ __launch_bounds__(256) void cvt_transpose_bf16(const float* __restrict__ src,
                                                          unsigned short* __restrict__ dst,
                                                          int R, int C) {
  __shared__ unsigned short tile[32][33];
  const int bc = blockIdx.x * 32;
  const int br = blockIdx.y * 32;
  const int tx = threadIdx.x & 31, ty = threadIdx.x >> 5;   // 32 x 8
  for (int i = ty; i < 32; i += 8)
    tile[i][tx] = f32_to_bf16(src[(size_t)(br + i) * C + bc + tx]);
  __syncthreads();
  for (int i = ty; i < 32; i += 8)
    dst[(size_t)(bc + i) * R + br + tx] = tile[tx][i];
}

// -------------------------- u16 [R][C] -> [C][R], batched over blockIdx.z
__global__ __launch_bounds__(256) void transpose_u16(const unsigned short* __restrict__ src,
                                                     unsigned short* __restrict__ dst,
                                                     int R, int C) {
  __shared__ unsigned short tile[32][33];
  const size_t base = (size_t)blockIdx.z * R * C;
  const int bc = blockIdx.x * 32;
  const int br = blockIdx.y * 32;
  const int tx = threadIdx.x & 31, ty = threadIdx.x >> 5;
  for (int i = ty; i < 32; i += 8)
    tile[i][tx] = src[base + (size_t)(br + i) * C + bc + tx];
  __syncthreads();
  for (int i = ty; i < 32; i += 8)
    dst[base + (size_t)(bc + i) * R + br + tx] = tile[tx][i];
}

// --------------------------------------------- bf16 GEMM (B pre-transposed)
// C[M,N] = A[M,K] * Bt[N,K]^T (+ bias[N]).  Block tile 128x128, BK=32.
// Async-DMA double-buffered LDS staging; 8 waves, each 32x64 via 2x4 WMMA.
__global__ __launch_bounds__(256) void gemm_bf16_tn(const unsigned short* __restrict__ A,
                                                    const unsigned short* __restrict__ Bt,
                                                    const float* __restrict__ bias,
                                                    float* __restrict__ C,
                                                    int M, int N, int K) {
  __shared__ __align__(16) unsigned short Asm[2][128 * 32];
  __shared__ __align__(16) unsigned short Bsm[2][128 * 32];

  const int tid  = threadIdx.x;
  const int wid  = tid >> 5;
  const int lane = tid & 31;
  const int lo   = lane & 15;
  const int hi   = lane >> 4;
  const int wave_m = wid >> 1;
  const int wave_n = wid & 1;
  const int blockRow = blockIdx.y * 128;
  const int blockCol = blockIdx.x * 128;

  v8f acc[2][4];
  for (int tm = 0; tm < 2; tm++)
    for (int tn = 0; tn < 4; tn++)
      for (int r = 0; r < 8; r++) acc[tm][tn][r] = 0.0f;

  const int nkt = K >> 5;

  // prologue: DMA tile 0 into buffer 0
  {
    for (int j = 0; j < 2; j++) {
      int chunk = tid + j * 256;
      int r = chunk >> 2, c = (chunk & 3) * 8;
      async_load_b128(A + (size_t)(blockRow + r) * K + c, &Asm[0][r * 32 + c]);
      async_load_b128(Bt + (size_t)(blockCol + r) * K + c, &Bsm[0][r * 32 + c]);
    }
  }

  for (int kt = 0; kt < nkt; kt++) {
    const int buf = kt & 1;
    wait_async0();       // our DMAs for buf are done
    __syncthreads();     // everyone's done; next buffer's readers finished too

    if (kt + 1 < nkt) {  // overlap: DMA next tile while computing this one
      const int nb = (kt + 1) & 1;
      const int ko = (kt + 1) * 32;
      for (int j = 0; j < 2; j++) {
        int chunk = tid + j * 256;
        int r = chunk >> 2, c = (chunk & 3) * 8;
        async_load_b128(A + (size_t)(blockRow + r) * K + ko + c,
                        &Asm[nb][r * 32 + c]);
        async_load_b128(Bt + (size_t)(blockCol + r) * K + ko + c,
                        &Bsm[nb][r * 32 + c]);
      }
    }

    FragBF fa[2], fb[4];
    for (int tm = 0; tm < 2; tm++) {
      const unsigned int* ap =
          (const unsigned int*)(&Asm[buf][(wave_m * 32 + tm * 16 + lo) * 32]);
      for (int p = 0; p < 8; p++) fa[tm].d[p] = ap[kdw(lane, p)];
    }
    for (int tn = 0; tn < 4; tn++) {
      const unsigned int* bp =
          (const unsigned int*)(&Bsm[buf][(wave_n * 64 + tn * 16 + lo) * 32]);
      for (int p = 0; p < 8; p++) fb[tn].d[p] = bp[kdw(lane, p)];
    }
    for (int tn = 0; tn < 4; tn++)
      for (int tm = 0; tm < 2; tm++)
        acc[tm][tn] = wmma_bf16(fa[tm].v, fb[tn].v, acc[tm][tn]);
  }

  for (int tm = 0; tm < 2; tm++)
    for (int tn = 0; tn < 4; tn++) {
      int row0 = blockRow + wave_m * 32 + tm * 16 + 8 * hi;
      int col  = blockCol + wave_n * 64 + tn * 16 + lo;
      float bv = bias ? bias[col] : 0.0f;
      for (int r = 0; r < 8; r++)
        C[(size_t)(row0 + r) * N + col] = acc[tm][tn][r] + bv;
    }
}

// ------------------------------------- fused RMSNorm + RoPE + per-head split
__global__ __launch_bounds__(256) void normrope(const float* __restrict__ qkv,
                                                const float* __restrict__ qnw,
                                                const float* __restrict__ knw,
                                                const long long* __restrict__ pos,
                                                unsigned short* __restrict__ qb,
                                                unsigned short* __restrict__ kb,
                                                unsigned short* __restrict__ vb) {
  const int t = blockIdx.x;
  const float* row = qkv + (size_t)t * QKV_N;
  const int tid = threadIdx.x, wid = tid >> 5, lane = tid & 31;

  float sq = 0.f, sk = 0.f;
  for (int i = tid; i < HID; i += 256) {
    float a = row[i];        sq += a * a;
    float b = row[HID + i];  sk += b * b;
  }
  for (int m = 1; m < 32; m <<= 1) {
    sq += __shfl_xor(sq, m);
    sk += __shfl_xor(sk, m);
  }
  __shared__ float rq[8], rk[8];
  if (lane == 0) { rq[wid] = sq; rk[wid] = sk; }
  __syncthreads();
  sq = 0.f; sk = 0.f;
  for (int i = 0; i < 8; i++) { sq += rq[i]; sk += rk[i]; }
  const float invq = rsqrtf(sq / (float)HID + 1e-6f);
  const float invk = rsqrtf(sk / (float)HID + 1e-6f);
  const float fp = (float)pos[t];
  const float LOG_THETA = 13.815510557964274f;  // ln(1e6)

  for (int idx = tid; idx < NH * 64; idx += 256) {
    int h = idx >> 6, j = idx & 63;
    float invf = __expf(-(float)j * (LOG_THETA / 64.0f));
    float c, s;
    __sincosf(fp * invf, &s, &c);
    int b0 = h * HD + j;
    size_t ob = ((size_t)h * T_SEQ + t) * HD + j;
    float q1 = row[b0] * invq * qnw[b0];
    float q2 = row[b0 + 64] * invq * qnw[b0 + 64];
    qb[ob]      = f32_to_bf16(q1 * c - q2 * s);
    qb[ob + 64] = f32_to_bf16(q2 * c + q1 * s);
    float k1 = row[HID + b0] * invk * knw[b0];
    float k2 = row[HID + b0 + 64] * invk * knw[b0 + 64];
    kb[ob]      = f32_to_bf16(k1 * c - k2 * s);
    kb[ob + 64] = f32_to_bf16(k2 * c + k1 * s);
  }
  for (int idx = tid; idx < HID; idx += 256) {
    int h = idx >> 7, d = idx & 127;
    vb[((size_t)h * T_SEQ + t) * HD + d] = f32_to_bf16(row[2 * HID + idx]);
  }
}

// ---------------------------------------- flash-style causal WMMA attention
// grid: (T/64, NH); 4 waves, each owns 16 query rows; async double-buffered
// K/V tiles of 32 keys in LDS.  V pre-transposed per head: Vt[h][d][T].
__global__ __launch_bounds__(128) void attn_wmma(const unsigned short* __restrict__ Q,
                                                 const unsigned short* __restrict__ Kmat,
                                                 const unsigned short* __restrict__ Vt,
                                                 unsigned short* __restrict__ O) {
  __shared__ __align__(16) unsigned short Ksh[2][32 * HD];    // [key][d]
  __shared__ __align__(16) unsigned short VshT[2][HD * 32];   // [d][key]
  __shared__ __align__(16) unsigned short Psh[4][16 * 32];

  const int h  = blockIdx.y;
  const int q0 = blockIdx.x * 64;
  const int tid = threadIdx.x, wid = tid >> 5, lane = tid & 31;
  const int lo = lane & 15, hi = lane >> 4;
  const float SCALE = 0.08838834764831845f;  // 1/sqrt(128)

  FragBF qf[4];
  {
    const unsigned int* qr = (const unsigned int*)
        (Q + ((size_t)h * T_SEQ + q0 + wid * 16 + lo) * HD);
    for (int kc = 0; kc < 4; kc++)
      for (int p = 0; p < 8; p++)
        qf[kc].d[p] = qr[kc * 16 + kdw(lane, p)];
  }

  float m[8], l[8];
  v8f o[8];
  for (int r = 0; r < 8; r++) { m[r] = -3.0e38f; l[r] = 0.0f; }
  for (int nt = 0; nt < 8; nt++)
    for (int r = 0; r < 8; r++) o[nt][r] = 0.0f;

  const int nkt = (q0 + 64) >> 5;   // causal: keys <= q0+63

  // prologue: DMA tile 0 into buffer 0
  for (int i = 0; i < 4; i++) {
    int chunk = tid + i * 128;                // 0..511
    int kr = chunk >> 4, kc = (chunk & 15) * 8;
    async_load_b128(Kmat + ((size_t)h * T_SEQ + kr) * HD + kc,
                    &Ksh[0][kr * HD + kc]);
    int vr = chunk >> 2, vc = (chunk & 3) * 8;
    async_load_b128(Vt + ((size_t)h * HD + vr) * T_SEQ + vc,
                    &VshT[0][vr * 32 + vc]);
  }

  for (int kt = 0; kt < nkt; kt++) {
    const int kbase = kt * 32;
    const int buf = kt & 1;
    wait_async0();
    __syncthreads();

    if (kt + 1 < nkt) {    // overlap next K/V tile DMA with compute
      const int nb = (kt + 1) & 1;
      const int ko = (kt + 1) * 32;
      for (int i = 0; i < 4; i++) {
        int chunk = tid + i * 128;
        int kr = chunk >> 4, kc = (chunk & 15) * 8;
        async_load_b128(Kmat + ((size_t)h * T_SEQ + ko + kr) * HD + kc,
                        &Ksh[nb][kr * HD + kc]);
        int vr = chunk >> 2, vc = (chunk & 3) * 8;
        async_load_b128(Vt + ((size_t)h * HD + vr) * T_SEQ + ko + vc,
                        &VshT[nb][vr * 32 + vc]);
      }
    }

    // S = Q K^T  (two 16x16 halves of the 16x32 score tile)
    float sv[2][8];
    for (int sh = 0; sh < 2; sh++) {
      v8f s;
      for (int r = 0; r < 8; r++) s[r] = 0.0f;
      const int col = sh * 16 + lo;
      const unsigned int* kr = (const unsigned int*)(&Ksh[buf][col * HD]);
      for (int kc = 0; kc < 4; kc++) {
        FragBF kf;
        for (int p = 0; p < 8; p++) kf.d[p] = kr[kc * 16 + kdw(lane, p)];
        s = wmma_bf16(qf[kc].v, kf.v, s);
      }
      const int key = kbase + col;
      const int qrow0 = q0 + wid * 16 + 8 * hi;
      for (int r = 0; r < 8; r++) {
        float x = s[r] * SCALE;
        if (key > qrow0 + r) x = -3.0e38f;    // causal mask
        sv[sh][r] = x;
      }
    }

    // online softmax (row reductions across the 16-lane C/D groups)
    float p0[8], p1[8];
    for (int r = 0; r < 8; r++) {
      float mx = fmaxf(sv[0][r], sv[1][r]);
      for (int msk = 1; msk < 16; msk <<= 1) mx = fmaxf(mx, __shfl_xor(mx, msk));
      float mnew = fmaxf(m[r], mx);
      float corr = __expf(m[r] - mnew);
      m[r] = mnew;
      float e0 = __expf(sv[0][r] - mnew);
      float e1 = __expf(sv[1][r] - mnew);
      float rs = e0 + e1;
      for (int msk = 1; msk < 16; msk <<= 1) rs += __shfl_xor(rs, msk);
      l[r] = l[r] * corr + rs;
      p0[r] = e0; p1[r] = e1;
      for (int nt = 0; nt < 8; nt++) o[nt][r] *= corr;
    }

    // lay P out row-major in LDS (D-layout -> A-layout transpose)
    unsigned short* P = Psh[wid];
    for (int r = 0; r < 8; r++) {
      int rw = r + 8 * hi;
      P[rw * 32 + lo]      = f32_to_bf16(p0[r]);
      P[rw * 32 + 16 + lo] = f32_to_bf16(p1[r]);
    }

    // O += P V
    FragBF pf;
    const unsigned int* pr = (const unsigned int*)(P + lo * 32);
    for (int p = 0; p < 8; p++) pf.d[p] = pr[kdw(lane, p)];
    for (int nt = 0; nt < 8; nt++) {
      FragBF vf;
      const unsigned int* vp =
          (const unsigned int*)(&VshT[buf][(nt * 16 + lo) * 32]);
      for (int p = 0; p < 8; p++) vf.d[p] = vp[kdw(lane, p)];
      o[nt] = wmma_bf16(pf.v, vf.v, o[nt]);
    }
  }

  // normalize and emit [t, h*128+d] bf16 for the output GEMM
  for (int nt = 0; nt < 8; nt++) {
    const int d = nt * 16 + lo;
    for (int r = 0; r < 8; r++) {
      int t = q0 + wid * 16 + 8 * hi + r;
      O[(size_t)t * HID + h * HD + d] = f32_to_bf16(o[nt][r] / l[r]);
    }
  }
}

// ---------------------------------------------------------------------------
extern "C" void kernel_launch(void* const* d_in, const int* in_sizes, int n_in,
                              void* d_out, int out_size, void* d_ws, size_t ws_size,
                              hipStream_t stream) {
  const float* hs       = (const float*)d_in[0];
  const float* wqkv     = (const float*)d_in[1];
  const float* bqkv     = (const float*)d_in[2];
  const float* qnw      = (const float*)d_in[3];
  const float* knw      = (const float*)d_in[4];
  const float* wo       = (const float*)d_in[5];
  const long long* pos  = (const long long*)d_in[6];
  float* out            = (float*)d_out;

  unsigned char* ws = (unsigned char*)d_ws;
  size_t off = 0;
  unsigned short* hsb    = (unsigned short*)(ws + off); off += (size_t)T_SEQ * HID * 2;
  unsigned short* wqkvT  = (unsigned short*)(ws + off); off += (size_t)HID * QKV_N * 2;
  unsigned short* woT    = (unsigned short*)(ws + off); off += (size_t)HID * HID * 2;
  float*          qkvf   = (float*)(ws + off);          off += (size_t)T_SEQ * QKV_N * 4;
  unsigned short* qb     = (unsigned short*)(ws + off); off += (size_t)T_SEQ * HID * 2;
  unsigned short* kb     = (unsigned short*)(ws + off); off += (size_t)T_SEQ * HID * 2;
  unsigned short* vb     = (unsigned short*)(ws + off); off += (size_t)T_SEQ * HID * 2;
  unsigned short* vbT    = (unsigned short*)(ws + off); off += (size_t)T_SEQ * HID * 2;
  unsigned short* attnb  = (unsigned short*)(ws + off); off += (size_t)T_SEQ * HID * 2;

  cvt_bf16<<<8192, 256, 0, stream>>>(hs, hsb, T_SEQ * HID);
  cvt_transpose_bf16<<<dim3(QKV_N / 32, HID / 32), 256, 0, stream>>>(
      wqkv, wqkvT, HID, QKV_N);
  cvt_transpose_bf16<<<dim3(HID / 32, HID / 32), 256, 0, stream>>>(
      wo, woT, HID, HID);

  gemm_bf16_tn<<<dim3(QKV_N / 128, T_SEQ / 128), 256, 0, stream>>>(
      hsb, wqkvT, bqkv, qkvf, T_SEQ, QKV_N, HID);

  normrope<<<T_SEQ, 256, 0, stream>>>(qkvf, qnw, knw, pos, qb, kb, vb);

  transpose_u16<<<dim3(HD / 32, T_SEQ / 32, NH), 256, 0, stream>>>(
      vb, vbT, T_SEQ, HD);

  attn_wmma<<<dim3(T_SEQ / 64, NH), 128, 0, stream>>>(qb, kb, vbT, attnb);

  gemm_bf16_tn<<<dim3(HID / 128, T_SEQ / 128), 256, 0, stream>>>(
      attnb, woT, nullptr, out, T_SEQ, HID, HID);
}
